// MPNEncoder_4252017623664
// MI455X (gfx1250) — matrix operand order repaired and checked by reading, not compiled, not run on hardware
//
#include <hip/hip_runtime.h>
#include <hip/hip_bf16.h>

// MPN GRU encoder for gfx1250 (MI455X).
// - All matmuls via v_wmma_f32_16x16x32_bf16 (f32 accumulate).
// - fmess-dependent GEMM parts precomputed once (loop-invariant across DEPTH=3).
// - Global->LDS staging (neighbor gather, fmess/fnode tiles) uses CDNA5
//   GLOBAL_LOAD_ASYNC_TO_LDS_B128 (ASYNCcnt) via inline asm.

#define N_NODES 40000
#define NODE_FDIM 256
#define N_EDGES 80000
#define INPUT_SIZE 384
#define HIDDEN 128
#define MAX_NB 6

typedef __attribute__((ext_vector_type(16))) __bf16        v16bf;
typedef __attribute__((ext_vector_type(8)))  float         v8f;
typedef __attribute__((ext_vector_type(4)))  unsigned int  u32x4;

union FragU {
  v16bf v;
  u32x4 q[2];
  unsigned short u[16];
};

__device__ __forceinline__ unsigned short f2bf(float f) {
  unsigned int u = __float_as_uint(f);
  unsigned int r = u + 0x7FFFu + ((u >> 16) & 1u);   // round-to-nearest-even
  return (unsigned short)(r >> 16);
}
__device__ __forceinline__ float bf2f(unsigned short b) {
  return __uint_as_float(((unsigned int)b) << 16);
}
__device__ __forceinline__ float sigmoidf(float x) {
  return 1.0f / (1.0f + __expf(-x));
}

// Async copy of 16 bytes global -> LDS (GLOBAL_LOAD_ASYNC_TO_LDS_B128, GV mode).
// lds_off: wave-relative LDS byte address (low 32 bits of flat shared pointer,
// see ISA 10.2).
__device__ __forceinline__ void async_g2l_b128(unsigned lds_off,
                                               unsigned long long gaddr) {
  asm volatile("global_load_async_to_lds_b128 %0, %1, off"
               :: "v"(lds_off), "v"(gaddr)
               : "memory");
}
__device__ __forceinline__ void wait_asynccnt0() {
  asm volatile("s_wait_asynccnt 0" ::: "memory");
}

// A-fragment (16x32 bf16) from a row-major bf16 tile in LDS.
// lanes 0-15: row=lane, elems 0-7 -> K=k0+0..7,   elems 8-15 -> K=k0+16..23
// lanes 16-31: row=lane-16, elems 0-7 -> K=k0+8..15, elems 8-15 -> K=k0+24..31
__device__ __forceinline__ v16bf load_a(const unsigned short* tile, int stride,
                                        int k0, int lane) {
  int r  = lane & 15;
  int kb = k0 + ((lane >> 4) << 3);
  FragU f;
  f.q[0] = *(const u32x4*)(tile + r * stride + kb);
  f.q[1] = *(const u32x4*)(tile + r * stride + kb + 16);
  return f.v;
}

// B-fragment from fragment-packed weights: ((nt*KT + kt)*32 + lane)*16 elems.
__device__ __forceinline__ v16bf load_b(const unsigned short* Bp, int nt, int kt,
                                        int KT, int lane) {
  const unsigned short* p = Bp + ((((size_t)nt * KT + kt) * 32 + lane) << 4);
  FragU f;
  f.q[0] = *(const u32x4*)(p);
  f.q[1] = *(const u32x4*)(p + 8);
  return f.v;
}

__device__ __forceinline__ v8f wmma_bf16(v16bf a, v16bf b, v8f c) {
  return __builtin_amdgcn_wmma_f32_16x16x32_bf16(false, a, false, b, (short)0, c,
                                                 false, false);
}

// ---------------------------------------------------------------- conversions
__global__ void k_f32_to_bf16(const float* __restrict__ src,
                              unsigned short* __restrict__ dst, int n) {
  int i = blockIdx.x * blockDim.x + threadIdx.x;
  int stride = gridDim.x * blockDim.x;
  for (; i < n; i += stride) dst[i] = f2bf(src[i]);
}

__global__ void k_zero_u16(unsigned short* __restrict__ dst, int n) {
  int i = blockIdx.x * blockDim.x + threadIdx.x;
  int stride = gridDim.x * blockDim.x;
  for (; i < n; i += stride) dst[i] = 0;
}

// Pack f32 weight W[K x 128] (rows r0..r0+KT*32) into WMMA B-fragment layout.
__global__ void k_swizzleB(const float* __restrict__ W, int r0, int KT,
                           unsigned short* __restrict__ dst) {
  int id = blockIdx.x * blockDim.x + threadIdx.x;
  int total = KT * 8 * 512;
  if (id >= total) return;
  int e    = id & 15;
  int lane = (id >> 4) & 31;
  int q    = id >> 9;
  int kt   = q % KT;
  int nt   = q / KT;
  int k = r0 + kt * 32 + ((lane >> 4) << 3) + (e & 7) + ((e >> 3) << 4);  // lanes16-31 hold K 8..15,24..31
  int n = nt * 16 + (lane & 15);
  dst[id] = f2bf(W[(size_t)k * HIDDEN + n]);
}

// ---------------------------------------------------- precompute mz / mr / mh
// mz = fmess@Wz[:384]+bz ; mr = fmess@Wr ; mh = fmess@Wh[:384]+bh
__global__ void __launch_bounds__(256) k_pre_gemm(
    const unsigned short* __restrict__ fmess_bf,
    const unsigned short* __restrict__ Bz1,
    const unsigned short* __restrict__ Br1,
    const unsigned short* __restrict__ Bh1,
    const float* __restrict__ bz, const float* __restrict__ bh,
    float* __restrict__ mz, float* __restrict__ mr, float* __restrict__ mh) {
  __shared__ unsigned short Atile[16][INPUT_SIZE];
  int e0 = blockIdx.x * 16;
  int t  = threadIdx.x;
  // async stage 16x384 bf16 = 768 x 16B chunks (3 per thread), global -> LDS
  for (int i = 0; i < 3; ++i) {
    int f  = t + i * 256;          // 0..767 ; 48 chunks per row
    int m  = f / 48;
    int c8 = (f % 48) * 8;         // u16 units, 16-byte chunks
    async_g2l_b128((unsigned)(size_t)&Atile[m][c8],
                   (unsigned long long)(fmess_bf +
                                        (size_t)(e0 + m) * INPUT_SIZE + c8));
  }
  wait_asynccnt0();
  __syncthreads();
  int wave = t >> 5, lane = t & 31;
  int nloc = lane & 15;
  int mo   = (lane >> 4) * 8;
  for (int it = 0; it < 3; ++it) {
    int tidx = wave + it * 8;          // 0..23 -> 3 output matrices x 8 n-tiles
    int sel  = tidx >> 3;
    int ntl  = tidx & 7;
    int n    = ntl * 16 + nloc;
    const unsigned short* B = (sel == 0) ? Bz1 : ((sel == 1) ? Br1 : Bh1);
    float* out   = (sel == 0) ? mz : ((sel == 1) ? mr : mh);
    float  bias  = (sel == 0) ? bz[n] : ((sel == 2) ? bh[n] : 0.0f);
    v8f acc;
    for (int v = 0; v < 8; ++v) acc[v] = bias;
    for (int ks = 0; ks < 12; ++ks) {
      v16bf a = load_a(&Atile[0][0], INPUT_SIZE, ks * 32, lane);
      v16bf b = load_b(B, ntl, ks, 12, lane);
      acc = wmma_bf16(a, b, acc);
    }
    for (int v = 0; v < 8; ++v)
      out[(size_t)(e0 + mo + v) * HIDDEN + n] = acc[v];
  }
}

// ------------------------------------------------------------------ GRU step
__global__ void __launch_bounds__(256) k_gru_step(
    const unsigned short* __restrict__ h_in,   // [E,128] bf16
    unsigned short* __restrict__ h_out,        // [E,128] bf16
    float* __restrict__ h_out_f32, int writeF32,
    const long long* __restrict__ bgraph,      // [E,6] int64
    const float* __restrict__ mz, const float* __restrict__ mr,
    const float* __restrict__ mh,
    const unsigned short* __restrict__ Bz2,
    const unsigned short* __restrict__ Bur,
    const unsigned short* __restrict__ Bh2,
    const float* __restrict__ bur) {
  __shared__ unsigned short hnei[MAX_NB][16][HIDDEN];   // 24 KB
  __shared__ float          sumh_f[16][HIDDEN];         // 8 KB
  __shared__ unsigned short sumh_bf[16][HIDDEN];        // 4 KB
  __shared__ unsigned short sg_bf[16][HIDDEN];          // 4 KB
  int e0 = blockIdx.x * 16;
  int t  = threadIdx.x;
  int wave = t >> 5, lane = t & 31;
  // async gather of neighbor rows: 96 (edge,nb) pairs x 256B.
  // Each wave instruction covers 2 pairs (half-wave per pair, 16B per lane).
  {
    int half = lane >> 4;            // 0/1 -> pair select
    int c    = (lane & 15) * 8;      // u16 units, 16B chunk within row
    for (int i = 0; i < 6; ++i) {
      int p = wave * 2 + half + i * 16;      // 0..95
      int m = p / MAX_NB, k = p % MAX_NB;
      long long idx = bgraph[(size_t)(e0 + m) * MAX_NB + k];
      async_g2l_b128((unsigned)(size_t)&hnei[k][m][c],
                     (unsigned long long)(h_in + (size_t)idx * HIDDEN + c));
    }
  }
  wait_asynccnt0();
  __syncthreads();
  // sum over neighbors (f32), plus bf16 copy for WMMA A
  for (int i = 0; i < 8; ++i) {
    int f = t + i * 256;
    int m = f >> 7, c = f & 127;
    float s = 0.f;
    for (int k = 0; k < MAX_NB; ++k) s += bf2f(hnei[k][m][c]);
    sumh_f[m][c]  = s;
    sumh_bf[m][c] = f2bf(s);
  }
  __syncthreads();
  int nloc = lane & 15, mo = (lane >> 4) * 8;
  int nt = wave;
  int n  = nt * 16 + nloc;
  // z = sigmoid(mz + sum_h @ Wz2)
  v8f zacc;
  for (int v = 0; v < 8; ++v) zacc[v] = mz[(size_t)(e0 + mo + v) * HIDDEN + n];
  for (int ks = 0; ks < 4; ++ks) {
    v16bf a = load_a(&sumh_bf[0][0], HIDDEN, ks * 32, lane);
    v16bf b = load_b(Bz2, nt, ks, 4, lane);
    zacc = wmma_bf16(a, b, zacc);
  }
  float z[8], sh[8], sg[8];
  for (int v = 0; v < 8; ++v) {
    z[v]  = sigmoidf(zacc[v]);
    sh[v] = sumh_f[mo + v][n];
    sg[v] = 0.f;
  }
  // hoist loop-invariant Ur B-fragments and (mr + bur) C-base out of nb loop
  v16bf burf[4];
  for (int ks = 0; ks < 4; ++ks) burf[ks] = load_b(Bur, nt, ks, 4, lane);
  float burn = bur[n];
  float rbase[8];
  for (int v = 0; v < 8; ++v)
    rbase[v] = mr[(size_t)(e0 + mo + v) * HIDDEN + n] + burn;
  // r_k = sigmoid(mr + h_nei_k @ Ur + bur); sum_gated += r_k * h_nei_k
  for (int k = 0; k < MAX_NB; ++k) {
    v8f racc;
    for (int v = 0; v < 8; ++v) racc[v] = rbase[v];
    for (int ks = 0; ks < 4; ++ks) {
      v16bf a = load_a(&hnei[k][0][0], HIDDEN, ks * 32, lane);
      racc = wmma_bf16(a, burf[ks], racc);
    }
    for (int v = 0; v < 8; ++v)
      sg[v] += sigmoidf(racc[v]) * bf2f(hnei[k][mo + v][n]);
  }
  for (int v = 0; v < 8; ++v) sg_bf[mo + v][n] = f2bf(sg[v]);
  __syncthreads();
  // pre_h = tanh(mh + sum_gated @ Wh2); h = (1-z)*sum_h + z*pre_h
  v8f pacc;
  for (int v = 0; v < 8; ++v) pacc[v] = mh[(size_t)(e0 + mo + v) * HIDDEN + n];
  for (int ks = 0; ks < 4; ++ks) {
    v16bf a = load_a(&sg_bf[0][0], HIDDEN, ks * 32, lane);
    v16bf b = load_b(Bh2, nt, ks, 4, lane);
    pacc = wmma_bf16(a, b, pacc);
  }
  for (int v = 0; v < 8; ++v) {
    float ph = tanhf(pacc[v]);
    float hv = (1.f - z[v]) * sh[v] + z[v] * ph;
    int e = e0 + mo + v;
    if (e == 0) hv = 0.f;                        // padding-row mask
    h_out[(size_t)e * HIDDEN + n] = f2bf(hv);
    if (writeF32) h_out_f32[(size_t)e * HIDDEN + n] = hv;
  }
}

// --------------------------------------------------------------- output head
__global__ void __launch_bounds__(256) k_out_head(
    const unsigned short* __restrict__ fnode_bf,  // [N,256] bf16
    const unsigned short* __restrict__ h_bf,      // [E,128] bf16
    const long long* __restrict__ agraph,         // [N,6] int64
    const unsigned short* __restrict__ Bo1,
    const unsigned short* __restrict__ Bo2,
    const float* __restrict__ bo,
    float* __restrict__ out) {
  __shared__ unsigned short fn_bf[16][NODE_FDIM];  // 8 KB
  __shared__ unsigned short nei_bf[16][HIDDEN];    // 4 KB
  int n0 = blockIdx.x * 16;
  int t  = threadIdx.x;
  // async stage fnode tile: 16x256 bf16 = 512 x 16B chunks (2 per thread)
  for (int i = 0; i < 2; ++i) {
    int f  = t + i * 256;          // 0..511 ; 32 chunks per row
    int m  = f >> 5;
    int c8 = (f & 31) * 8;
    async_g2l_b128((unsigned)(size_t)&fn_bf[m][c8],
                   (unsigned long long)(fnode_bf +
                                        (size_t)(n0 + m) * NODE_FDIM + c8));
  }
  for (int i = 0; i < 8; ++i) {                    // gather + sum h[agraph]
    int f = t + i * 256;
    int m = f >> 7, c = f & 127;
    float s = 0.f;
    for (int k = 0; k < MAX_NB; ++k) {
      long long idx = agraph[(size_t)(n0 + m) * MAX_NB + k];
      s += bf2f(h_bf[(size_t)idx * HIDDEN + c]);
    }
    nei_bf[m][c] = f2bf(s);
  }
  wait_asynccnt0();
  __syncthreads();
  int wave = t >> 5, lane = t & 31;
  int nloc = lane & 15, mo = (lane >> 4) * 8;
  int n = wave * 16 + nloc;
  v8f acc;
  float b0 = bo[n];
  for (int v = 0; v < 8; ++v) acc[v] = b0;
  for (int ks = 0; ks < 8; ++ks) {                 // fnode @ Wo[:256]
    v16bf a = load_a(&fn_bf[0][0], NODE_FDIM, ks * 32, lane);
    v16bf b = load_b(Bo1, wave, ks, 8, lane);
    acc = wmma_bf16(a, b, acc);
  }
  for (int ks = 0; ks < 4; ++ks) {                 // nei_message @ Wo[256:]
    v16bf a = load_a(&nei_bf[0][0], HIDDEN, ks * 32, lane);
    v16bf b = load_b(Bo2, wave, ks, 4, lane);
    acc = wmma_bf16(a, b, acc);
  }
  for (int v = 0; v < 8; ++v) {
    float x = acc[v];
    x = x > 0.f ? x : 0.f;                         // ReLU
    int node = n0 + mo + v;
    if (node == 0) x = 0.f;                        // padding-row mask
    out[(size_t)node * HIDDEN + n] = x;
  }
}

// ------------------------------------------------------------------- launch
extern "C" void kernel_launch(void* const* d_in, const int* in_sizes, int n_in,
                              void* d_out, int out_size, void* d_ws, size_t ws_size,
                              hipStream_t stream) {
  (void)in_sizes; (void)n_in; (void)out_size; (void)ws_size;
  const float*     fnode  = (const float*)d_in[0];
  const float*     fmess  = (const float*)d_in[1];
  const long long* agraph = (const long long*)d_in[2];
  const long long* bgraph = (const long long*)d_in[3];
  const float*     Wz     = (const float*)d_in[4];
  const float*     bz     = (const float*)d_in[5];
  const float*     Wr     = (const float*)d_in[6];
  const float*     Ur     = (const float*)d_in[7];
  const float*     bur    = (const float*)d_in[8];
  const float*     Wh     = (const float*)d_in[9];
  const float*     bh     = (const float*)d_in[10];
  const float*     Wo     = (const float*)d_in[11];
  const float*     bo     = (const float*)d_in[12];

  const int E = N_EDGES, Nn = N_NODES;
  char* ws = (char*)d_ws;
  size_t off = 0;
  auto ALLOC = [&](size_t nbytes) -> void* {
    void* p = (void*)(ws + off);
    off += (nbytes + 255) & ~(size_t)255;
    return p;
  };
  unsigned short* fmess_bf = (unsigned short*)ALLOC((size_t)E * INPUT_SIZE * 2);
  unsigned short* fnode_bf = (unsigned short*)ALLOC((size_t)Nn * NODE_FDIM * 2);
  float* mz = (float*)ALLOC((size_t)E * HIDDEN * 4);
  float* mr = (float*)ALLOC((size_t)E * HIDDEN * 4);
  float* mh = (float*)ALLOC((size_t)E * HIDDEN * 4);
  unsigned short* hA = (unsigned short*)ALLOC((size_t)E * HIDDEN * 2);
  unsigned short* hB = (unsigned short*)ALLOC((size_t)E * HIDDEN * 2);
  unsigned short* Bz1 = (unsigned short*)ALLOC((size_t)12 * 8 * 512 * 2);
  unsigned short* Br1 = (unsigned short*)ALLOC((size_t)12 * 8 * 512 * 2);
  unsigned short* Bh1 = (unsigned short*)ALLOC((size_t)12 * 8 * 512 * 2);
  unsigned short* Bz2 = (unsigned short*)ALLOC((size_t)4 * 8 * 512 * 2);
  unsigned short* Bur = (unsigned short*)ALLOC((size_t)4 * 8 * 512 * 2);
  unsigned short* Bh2 = (unsigned short*)ALLOC((size_t)4 * 8 * 512 * 2);
  unsigned short* Bo1 = (unsigned short*)ALLOC((size_t)8 * 8 * 512 * 2);
  unsigned short* Bo2 = (unsigned short*)ALLOC((size_t)4 * 8 * 512 * 2);

  const int THR = 256;
  k_f32_to_bf16<<<2048, THR, 0, stream>>>(fmess, fmess_bf, E * INPUT_SIZE);
  k_f32_to_bf16<<<2048, THR, 0, stream>>>(fnode, fnode_bf, Nn * NODE_FDIM);
  k_swizzleB<<<(12 * 8 * 512 + THR - 1) / THR, THR, 0, stream>>>(Wz, 0, 12, Bz1);
  k_swizzleB<<<(4 * 8 * 512 + THR - 1) / THR, THR, 0, stream>>>(Wz, 384, 4, Bz2);
  k_swizzleB<<<(12 * 8 * 512 + THR - 1) / THR, THR, 0, stream>>>(Wr, 0, 12, Br1);
  k_swizzleB<<<(4 * 8 * 512 + THR - 1) / THR, THR, 0, stream>>>(Ur, 0, 4, Bur);
  k_swizzleB<<<(12 * 8 * 512 + THR - 1) / THR, THR, 0, stream>>>(Wh, 0, 12, Bh1);
  k_swizzleB<<<(4 * 8 * 512 + THR - 1) / THR, THR, 0, stream>>>(Wh, 384, 4, Bh2);
  k_swizzleB<<<(8 * 8 * 512 + THR - 1) / THR, THR, 0, stream>>>(Wo, 0, 8, Bo1);
  k_swizzleB<<<(4 * 8 * 512 + THR - 1) / THR, THR, 0, stream>>>(Wo, 256, 4, Bo2);
  k_zero_u16<<<2048, THR, 0, stream>>>(hA, E * HIDDEN);
  k_pre_gemm<<<E / 16, THR, 0, stream>>>(fmess_bf, Bz1, Br1, Bh1, bz, bh,
                                         mz, mr, mh);
  float* h_f32_out = (float*)d_out + (size_t)Nn * HIDDEN;
  k_gru_step<<<E / 16, THR, 0, stream>>>(hA, hB, h_f32_out, 0, bgraph,
                                         mz, mr, mh, Bz2, Bur, Bh2, bur);
  k_gru_step<<<E / 16, THR, 0, stream>>>(hB, hA, h_f32_out, 0, bgraph,
                                         mz, mr, mh, Bz2, Bur, Bh2, bur);
  k_gru_step<<<E / 16, THR, 0, stream>>>(hA, hB, h_f32_out, 1, bgraph,
                                         mz, mr, mh, Bz2, Bur, Bh2, bur);
  k_out_head<<<Nn / 16, THR, 0, stream>>>(fnode_bf, hB, agraph, Bo1, Bo2, bo,
                                          (float*)d_out);
}
